// SSVEP_Classifier_83631603188243
// MI455X (gfx1250) — matrix operand order repaired and checked by reading, not compiled, not run on hardware
//
#include <hip/hip_runtime.h>
#include <hip/hip_bf16.h>
#include <math.h>

// ---------------------------------------------------------------------------
// Mamba-style SSM classifier for MI455X (gfx1250, wave32).
//   B=1024, L=250, D_MODEL=8, D_INNER=16, D_STATE=16, DT_RANK=1, NC=12
// Kernel 1: in_proj + causal conv + silu + x_proj + softplus(delta)  (LDS fused)
// Kernel 2: selective scan; thread = (batch, d); state in registers;
//           B/C double-buffered via GLOBAL_LOAD_ASYNC_TO_LDS (ASYNCcnt),
//           wave-local LDS reuse -> no block barriers in the 250-step loop.
// Kernel 3: final FC GEMM (1024 x 2000 @ 2000 x 12) on V_WMMA_F32_16X16X4_F32.
// ---------------------------------------------------------------------------

#define BATCH 1024
#define SEQ   250
#define DM    8      // D_MODEL
#define DI    16     // D_INNER
#define DS    16     // D_STATE
#define DCONV 4
#define NC    12     // NUM_CLASSES
#define KFC   2000   // D_MODEL * SEQ_LEN

typedef __attribute__((ext_vector_type(2)))  float    v2f;
typedef __attribute__((ext_vector_type(8)))  float    v8f;
typedef __attribute__((ext_vector_type(16))) _Float16 v16h;

#if __has_builtin(__builtin_amdgcn_wmma_f32_16x16x4_f32)
#define USE_WMMA_F32X4 1
#else
#define USE_WMMA_F32X4 0
#endif

#if __has_builtin(__builtin_amdgcn_global_load_async_to_lds_b32) && \
    __has_builtin(__builtin_amdgcn_s_wait_asynccnt)
#define HAVE_ASYNC_LDS 1
#else
#define HAVE_ASYNC_LDS 0
#endif

#if HAVE_ASYNC_LDS
// Direct global->LDS async copy (one b32 per lane). The builtin takes int*
// operands: global source (addrspace 1) and LDS destination (addrspace 3);
// the LDS pointer is formed from the low 32 bits of the generic address
// (generic LDS addr = SHARED_BASE | offset).
__device__ __forceinline__ void async_copy_b32(const float* g, float* l) {
    __builtin_amdgcn_global_load_async_to_lds_b32(
        (__attribute__((address_space(1))) int*)(uintptr_t)g,
        (__attribute__((address_space(3))) int*)(uint32_t)(uintptr_t)l,
        /*offset=*/0, /*cpol=*/0);
}
#endif

__device__ __forceinline__ float sigmoidf_(float v) { return 1.f / (1.f + __expf(-v)); }
__device__ __forceinline__ float siluf_(float v)    { return v * sigmoidf_(v); }
__device__ __forceinline__ float softplusf_(float v){ return (v > 20.f) ? v : log1pf(__expf(v)); }

// ---------------------------------------------------------------------------
// Kernel 1: per-batch front end. blockIdx.x = batch, threadIdx.x = timestep.
// ---------------------------------------------------------------------------
__global__ void __launch_bounds__(256) mamba_front(
    const float* __restrict__ x,      // (B, DM, SEQ)
    const float* __restrict__ w_in,   // (2*DI, DM)
    const float* __restrict__ conv_w, // (DI, DCONV)
    const float* __restrict__ conv_b, // (DI,)
    const float* __restrict__ w_xp,   // (1+2*DS, DI)
    const float* __restrict__ dt_w,   // (DI, 1)
    const float* __restrict__ dt_b,   // (DI,)
    float* __restrict__ o_u,          // (B, SEQ, DI)
    float* __restrict__ o_dl,         // (B, SEQ, DI)
    float* __restrict__ o_B,          // (B, SEQ, DS)
    float* __restrict__ o_C,          // (B, SEQ, DS)
    float* __restrict__ o_sz)         // (B, SEQ, DI)  silu(z)
{
    __shared__ float s_xin[SEQ * DI];          // 16000 B conv input tile
    __shared__ float s_win[2 * DI * DM];
    __shared__ float s_wxp[(1 + 2 * DS) * DI];
    __shared__ float s_cw[DI * DCONV];
    __shared__ float s_cb[DI];
    __shared__ float s_dtw[DI];
    __shared__ float s_dtb[DI];

    const int tid = threadIdx.x;
    const int b   = blockIdx.x;

    for (int i = tid; i < 2 * DI * DM; i += 256)        s_win[i] = w_in[i];
    for (int i = tid; i < (1 + 2 * DS) * DI; i += 256)  s_wxp[i] = w_xp[i];
    if (tid < DI * DCONV) s_cw[tid] = conv_w[tid];
    if (tid < DI) { s_cb[tid] = conv_b[tid]; s_dtw[tid] = dt_w[tid]; s_dtb[tid] = dt_b[tid]; }
    __syncthreads();

    const int t = tid;
    const size_t rowbase = ((size_t)b * SEQ + t) * DI;

    if (t < SEQ) {
        float xr[DM];
        #pragma unroll
        for (int j = 0; j < DM; ++j) xr[j] = x[((size_t)b * DM + j) * SEQ + t];
        #pragma unroll
        for (int c = 0; c < 2 * DI; ++c) {
            float acc = 0.f;
            #pragma unroll
            for (int j = 0; j < DM; ++j) acc = fmaf(xr[j], s_win[c * DM + j], acc);
            if (c < DI) s_xin[t * DI + c] = acc;                // x_in branch -> conv
            else        o_sz[rowbase + (c - DI)] = siluf_(acc); // z branch -> silu(z)
        }
    }
    __syncthreads();

    if (t < SEQ) {
        float u[DI];
        #pragma unroll
        for (int d = 0; d < DI; ++d) {
            float acc = s_cb[d];
            #pragma unroll
            for (int k = 0; k < DCONV; ++k) {
                int tt = t + k - (DCONV - 1);            // causal window
                if (tt >= 0) acc = fmaf(s_cw[d * DCONV + k], s_xin[tt * DI + d], acc);
            }
            u[d] = siluf_(acc);
            o_u[rowbase + d] = u[d];
        }
        // x_proj: row 0 -> dt (DT_RANK==1), rows 1..16 -> B, rows 17..32 -> C
        float dts = 0.f;
        #pragma unroll
        for (int j = 0; j < DI; ++j) dts = fmaf(u[j], s_wxp[j], dts);
        #pragma unroll
        for (int s = 0; s < DS; ++s) {
            float accB = 0.f, accC = 0.f;
            #pragma unroll
            for (int j = 0; j < DI; ++j) {
                accB = fmaf(u[j], s_wxp[(1 + s) * DI + j], accB);
                accC = fmaf(u[j], s_wxp[(1 + DS + s) * DI + j], accC);
            }
            o_B[rowbase + s] = accB;
            o_C[rowbase + s] = accC;
        }
        #pragma unroll
        for (int d = 0; d < DI; ++d)
            o_dl[rowbase + d] = softplusf_(fmaf(dts, s_dtw[d], s_dtb[d]));
    }
}

// ---------------------------------------------------------------------------
// Kernel 2: selective scan + fused out_proj. 256 thr = 16 batches x 16 chans.
// LDS rows sB/sC/sY[bl] are produced AND consumed by wave (bl/2) only, and
// CDNA5 keeps one wave's LDS ops in order -> no block barriers in the loop.
// Next timestep's B/C are double-buffered through the async global->LDS path
// (ASYNCcnt) so their latency hides behind 16 exps + 32 FMAs of VALU work.
// ---------------------------------------------------------------------------
__global__ void __launch_bounds__(256) mamba_scan(
    const float* __restrict__ u,
    const float* __restrict__ dl,
    const float* __restrict__ Bm,
    const float* __restrict__ Cm,
    const float* __restrict__ sz,
    const float* __restrict__ A_log,  // (DI, DS)
    const float* __restrict__ Dv,     // (DI,)
    const float* __restrict__ w_out,  // (DM, DI)
    float* __restrict__ flat)         // (B, KFC), layout [b][t*DM + m]
{
    __shared__ float sB[2][16][DS];
    __shared__ float sC[2][16][DS];
    __shared__ float sY[16][DI];
    __shared__ float sW[DM * DI];

    const int tid = threadIdx.x;
    const int d   = tid & 15;        // channel
    const int bl  = tid >> 4;        // local batch
    const int b   = blockIdx.x * 16 + bl;

    if (tid < DM * DI) sW[tid] = w_out[tid];

    float a_s[DS];
    #pragma unroll
    for (int s = 0; s < DS; ++s) a_s[s] = -__expf(A_log[d * DS + s]);   // A = -exp(A_log)
    float h[DS];
    #pragma unroll
    for (int s = 0; s < DS; ++s) h[s] = 0.f;
    const float Dd = Dv[d];

    size_t idx = ((size_t)b * SEQ) * DI + d;

#if HAVE_ASYNC_LDS
    async_copy_b32(&Bm[idx], &sB[0][bl][d]);      // stage t = 0
    async_copy_b32(&Cm[idx], &sC[0][bl][d]);
#endif
    __syncthreads();                              // sW visibility (once)

    for (int t = 0; t < SEQ; ++t, idx += DI) {
        const int buf = t & 1;
        float dv  = dl[idx];
        float uv  = u[idx];
        float szv = sz[idx];
        if (t + 8 < SEQ) {                        // stream-ahead -> global_prefetch_b8
            __builtin_prefetch(&dl[idx + 8 * DI], 0, 0);
            __builtin_prefetch(&u[idx + 8 * DI], 0, 0);
        }
#if HAVE_ASYNC_LDS
        if (t + 1 < SEQ) {                        // issue t+1 into the other buffer
            async_copy_b32(&Bm[idx + DI], &sB[buf ^ 1][bl][d]);
            async_copy_b32(&Cm[idx + DI], &sC[buf ^ 1][bl][d]);
            __builtin_amdgcn_s_wait_asynccnt(2);  // t's pair (in-order) now resident
        } else {
            __builtin_amdgcn_s_wait_asynccnt(0);
        }
        asm volatile("" ::: "memory");            // keep ds loads below the wait
#else
        sB[buf][bl][d] = Bm[idx];
        sC[buf][bl][d] = Cm[idx];
#endif
        __builtin_amdgcn_wave_barrier();

        float y = 0.f;
        #pragma unroll
        for (int s = 0; s < DS; ++s) {
            float w = __expf(dv * a_s[s]);                 // dA
            h[s] = fmaf(w, h[s], dv * sB[buf][bl][s] * uv);// h = dA*h + dBu
            y    = fmaf(h[s], sC[buf][bl][s], y);          // y += h . C
        }
        y = fmaf(uv, Dd, y) * szv;                         // (+ u*D) * silu(z)
        sY[bl][d] = y;
        __builtin_amdgcn_wave_barrier();                   // same-wave LDS reuse

        if (d < DM) {                                      // fused out_proj (K=16)
            float o = 0.f;
            #pragma unroll
            for (int j = 0; j < DI; ++j) o = fmaf(sY[bl][j], sW[d * DI + j], o);
            flat[(size_t)b * KFC + t * DM + d] = o;
        }
        __builtin_amdgcn_wave_barrier();
    }
}

// ---------------------------------------------------------------------------
// Kernel 3: final FC GEMM on WMMA. One wave (32 lanes) per 16-row tile.
//   D(16x16) = A(16xK) * B(Kx16) + C, N padded 12 -> 16.
// ---------------------------------------------------------------------------
__global__ void __launch_bounds__(32) fc_wmma(
    const float* __restrict__ flat,   // (B, KFC)
    const float* __restrict__ fcw,    // (NC, KFC)
    const float* __restrict__ fcb,    // (NC,)
    float* __restrict__ out)          // (B, NC)
{
    const int lane = threadIdx.x;
    const int half = lane >> 4;       // 0: lanes 0-15, 1: lanes 16-31
    const int m    = lane & 15;       // row (A) / col (B,C)
    const int m0   = blockIdx.x * 16;
    const float bscale = (m < NC) ? 1.f : 0.f;     // branchless N padding
    const int   brow   = (m < NC) ? m : 0;

    v8f c = {};

#if USE_WMMA_F32X4
    // 32-bit A 16x4 layout: lanes 0-15 hold K=0,1 ; lanes 16-31 hold K=2,3.
    const float* ap = flat + (size_t)(m0 + m) * KFC + 2 * half;
    const float* bp = fcw  + (size_t)brow * KFC + 2 * half;
    for (int k0 = 0; k0 < KFC; k0 += 4) {
        v2f a, bv;
        a.x  = ap[k0];            a.y  = ap[k0 + 1];
        bv.x = bp[k0] * bscale;   bv.y = bp[k0 + 1] * bscale;
        c = __builtin_amdgcn_wmma_f32_16x16x4_f32(
                false, a, false, bv, (short)0, c, false, false);
    }
#else
    // Fallback: f16 inputs, f32 accumulate (codegen-confirmed builtin).
    const float* ap = flat + (size_t)(m0 + m) * KFC;
    const float* bp = fcw  + (size_t)brow * KFC;
    for (int k0 = 0; k0 < KFC; k0 += 32) {
        v16h a, bv;
        #pragma unroll
        for (int i = 0; i < 8; ++i) {           // A: K chunks [half*8, +8) U [16+half*8, +8)
            int ka  = k0 + half * 8 + i;
            int ka2 = k0 + 16 + half * 8 + i;
            a[i]     = (_Float16)((ka  < KFC) ? ap[ka]  : 0.f);
            a[8 + i] = (_Float16)((ka2 < KFC) ? ap[ka2] : 0.f);
        }
        #pragma unroll
        for (int i = 0; i < 16; ++i) {          // B: K chunks [half*8, +8) U [16+half*8, +8)
            int kb = k0 + ((i < 8) ? (half * 8 + i) : (16 + half * 8 + (i - 8)));
            bv[i] = (_Float16)((kb < KFC) ? bp[kb] * bscale : 0.f);
        }
        c = __builtin_amdgcn_wmma_f32_16x16x32_f16(
                false, a, false, bv, (short)0, c, false, false);
    }
#endif

    const float bias = (m < NC) ? fcb[m] : 0.f;
    #pragma unroll
    for (int j = 0; j < 8; ++j) {               // C/D: VGPR j = row j (+8 for upper half)
        int row = m0 + j + 8 * half;
        if (m < NC) out[row * NC + m] = c[j] + bias;
    }
}

// ---------------------------------------------------------------------------
extern "C" void kernel_launch(void* const* d_in, const int* in_sizes, int n_in,
                              void* d_out, int out_size, void* d_ws, size_t ws_size,
                              hipStream_t stream)
{
    (void)in_sizes; (void)n_in; (void)out_size; (void)ws_size;
    const float* x      = (const float*)d_in[0];
    const float* w_in   = (const float*)d_in[1];
    const float* conv_w = (const float*)d_in[2];
    const float* conv_b = (const float*)d_in[3];
    const float* w_xp   = (const float*)d_in[4];
    const float* dt_w   = (const float*)d_in[5];
    const float* dt_b   = (const float*)d_in[6];
    const float* A_log  = (const float*)d_in[7];
    const float* Dv     = (const float*)d_in[8];
    const float* w_out  = (const float*)d_in[9];
    const float* fcw    = (const float*)d_in[10];
    const float* fcb    = (const float*)d_in[11];
    float* out = (float*)d_out;

    // Workspace layout (floats): u | delta | B | C | silu(z) | flat  (~90 MB)
    float* W = (float*)d_ws;
    const size_t N1 = (size_t)BATCH * SEQ * DI;   // 4,096,000
    float* w_u  = W;
    float* w_dl = W + 1 * N1;
    float* w_B  = W + 2 * N1;
    float* w_C  = W + 3 * N1;
    float* w_sz = W + 4 * N1;
    float* w_fl = W + 5 * N1;                     // BATCH * KFC

    mamba_front<<<BATCH, 256, 0, stream>>>(x, w_in, conv_w, conv_b, w_xp, dt_w, dt_b,
                                           w_u, w_dl, w_B, w_C, w_sz);
    mamba_scan<<<BATCH / 16, 256, 0, stream>>>(w_u, w_dl, w_B, w_C, w_sz,
                                               A_log, Dv, w_out, w_fl);
    fc_wmma<<<BATCH / 16, 32, 0, stream>>>(w_fl, fcw, fcb, out);
}
